// ViMambaEncoder_77300821394077
// MI455X (gfx1250) — matrix operand order repaired and checked by reading, not compiled, not run on hardware
//
#include <hip/hip_runtime.h>
#include <hip/hip_bf16.h>
#include <stdint.h>

namespace {

constexpr int kEmbed  = 384;
constexpr int kDepth  = 24;
constexpr int kDin    = 768;
constexpr int kNState = 16;
constexpr int kDtRank = 24;
constexpr int kXdblPad = 64;          // 56 (=24+16+16) padded to 64
constexpr int kXdblReal = 56;
constexpr int kB = 16;
constexpr int kL = 196;
constexpr int kTok = kB * kL;         // 3136
constexpr int kD2 = 2 * kDin;         // 1536
constexpr float kEps = 1e-5f;

typedef __attribute__((ext_vector_type(16))) __bf16 v16bf;
typedef __attribute__((ext_vector_type(8)))  float  v8f;

__device__ __forceinline__ unsigned short f32_to_bf16(float f) {
  union { float f; unsigned u; } c; c.f = f;
  unsigned u = c.u;
  u += 0x7fffu + ((u >> 16) & 1u);   // round to nearest even
  return (unsigned short)(u >> 16);
}

union Frag16 {
  v16bf v;
  unsigned short u[16];
  uint4 q[2];
};

// ---------------------------------------------------------------- init / cvt
__global__ void init_kernel(const float* __restrict__ rgb, const float* __restrict__ dep,
                            float* __restrict__ h_rgb, float* __restrict__ h_dep,
                            float* __restrict__ r_rgb, float* __restrict__ r_dep, int n) {
  int i = blockIdx.x * blockDim.x + threadIdx.x;
  if (i >= n) return;
  h_rgb[i] = rgb[i]; h_dep[i] = dep[i];
  r_rgb[i] = 0.f;    r_dep[i] = 0.f;
}

__global__ void cvt_bf16_kernel(const float* __restrict__ in, unsigned short* __restrict__ out, int n) {
  int i = blockIdx.x * blockDim.x + threadIdx.x;
  if (i < n) out[i] = f32_to_bf16(in[i]);
}

// ------------------------------------------------- residual add + layernorm
__global__ void addnorm_kernel(const float* __restrict__ h, float* __restrict__ r,
                               const float* __restrict__ w, const float* __restrict__ b,
                               unsigned short* __restrict__ xn) {
  __shared__ float sred[128];
  int n = blockIdx.x;
  int t = threadIdx.x;       // 128 threads, 3 elements each (384)
  float v[3]; float s = 0.f;
#pragma unroll
  for (int i = 0; i < 3; ++i) {
    int c = t + i * 128;
    float x = h[(size_t)n * kEmbed + c] + r[(size_t)n * kEmbed + c];
    r[(size_t)n * kEmbed + c] = x;
    v[i] = x; s += x;
  }
  sred[t] = s; __syncthreads();
  for (int off = 64; off > 0; off >>= 1) { if (t < off) sred[t] += sred[t + off]; __syncthreads(); }
  float mu = sred[0] * (1.f / kEmbed);
  __syncthreads();
  s = 0.f;
#pragma unroll
  for (int i = 0; i < 3; ++i) { float d = v[i] - mu; s += d * d; }
  sred[t] = s; __syncthreads();
  for (int off = 64; off > 0; off >>= 1) { if (t < off) sred[t] += sred[t + off]; __syncthreads(); }
  float rs = rsqrtf(sred[0] * (1.f / kEmbed) + kEps);
#pragma unroll
  for (int i = 0; i < 3; ++i) {
    int c = t + i * 128;
    xn[(size_t)n * kEmbed + c] = f32_to_bf16((v[i] - mu) * rs * w[c] + b[c]);
  }
}

__global__ void final_norm_kernel(const float* __restrict__ h, const float* __restrict__ r,
                                  const float* __restrict__ w, const float* __restrict__ b,
                                  float* __restrict__ out) {
  __shared__ float sred[128];
  int n = blockIdx.x;
  int t = threadIdx.x;
  float v[3]; float s = 0.f;
#pragma unroll
  for (int i = 0; i < 3; ++i) {
    int c = t + i * 128;
    float x = h[(size_t)n * kEmbed + c] + r[(size_t)n * kEmbed + c];
    v[i] = x; s += x;
  }
  sred[t] = s; __syncthreads();
  for (int off = 64; off > 0; off >>= 1) { if (t < off) sred[t] += sred[t + off]; __syncthreads(); }
  float mu = sred[0] * (1.f / kEmbed);
  __syncthreads();
  s = 0.f;
#pragma unroll
  for (int i = 0; i < 3; ++i) { float d = v[i] - mu; s += d * d; }
  sred[t] = s; __syncthreads();
  for (int off = 64; off > 0; off >>= 1) { if (t < off) sred[t] += sred[t + off]; __syncthreads(); }
  float rs = rsqrtf(sred[0] * (1.f / kEmbed) + kEps);
#pragma unroll
  for (int i = 0; i < 3; ++i) {
    int c = t + i * 128;
    out[(size_t)n * kEmbed + c] = (v[i] - mu) * rs * w[c] + b[c];
  }
}

// --------------------------------------------------------------- WMMA GEMM
// C[M, ldc(f32)] = A[M,K(bf16)] x W[Nout,K(bf16)]^T
// One wave computes a 16(M) x 64(N) block: A fragment reused across 4 WMMAs.
// GUARD=false requires Nout == ldc (no column guard in the hot loop).
template <bool GUARD>
__global__ void wmma_gemm_kernel(const unsigned short* __restrict__ A,
                                 const unsigned short* __restrict__ Wbf,
                                 float* __restrict__ C,
                                 int M, int K, int Nout, int ldc) {
  int wid  = (int)((blockIdx.x * blockDim.x + threadIdx.x) >> 5);
  int lane = (int)(threadIdx.x & 31);
  int tilesN = ldc >> 6;                 // 64 output cols per wave
  int tilesM = M >> 4;
  if (wid >= tilesM * tilesN) return;    // wave-uniform
  int tm = wid / tilesN, tn = wid % tilesN;
  int m0 = tm << 4, n0 = tn << 6;
  int half = lane >> 4;
  int l16  = lane & 15;
  int mrow = m0 + l16;

  const unsigned short* Abase = A + (size_t)mrow * K;
  const unsigned short* Bb[4];
  bool nv[4];
#pragma unroll
  for (int t = 0; t < 4; ++t) {
    int ncol = n0 + t * 16 + l16;
    nv[t] = GUARD ? (ncol < Nout) : true;
    Bb[t] = Wbf + (size_t)ncol * K;
  }

  v8f acc0 = {}, acc1 = {}, acc2 = {}, acc3 = {};
  for (int k0 = 0; k0 < K; k0 += 32) {
    __builtin_prefetch((const void*)(Abase + k0 + 128), 0, 1);        // global_prefetch_b8
    __builtin_prefetch((const void*)(Bb[0] + k0 + 128), 0, 1);
    // A 16x32 bf16 fragment: u[0..7] = K[k0+half*8 ..], u[8..15] = K[k0+16+half*8 ..]
    Frag16 a;
    const unsigned short* ap = Abase + k0 + half * 8;
    a.q[0] = *(const uint4*)(ap);
    a.q[1] = *(const uint4*)(ap + 16);
    // 4 B fragments (32x16 each): lane column, u[0..7] = K[k0+half*8 ..], u[8..15] = K[k0+16+half*8 ..]
    Frag16 bf[4];
#pragma unroll
    for (int t = 0; t < 4; ++t) {
      if (nv[t]) {
        const unsigned short* bp = Bb[t] + k0 + half * 8;
        bf[t].q[0] = *(const uint4*)(bp);
        bf[t].q[1] = *(const uint4*)(bp + 16);
      } else {
        bf[t].q[0] = make_uint4(0u, 0u, 0u, 0u);
        bf[t].q[1] = make_uint4(0u, 0u, 0u, 0u);
      }
    }
    acc0 = __builtin_amdgcn_wmma_f32_16x16x32_bf16(false, a.v, false, bf[0].v, (short)0, acc0, false, false);
    acc1 = __builtin_amdgcn_wmma_f32_16x16x32_bf16(false, a.v, false, bf[1].v, (short)0, acc1, false, false);
    acc2 = __builtin_amdgcn_wmma_f32_16x16x32_bf16(false, a.v, false, bf[2].v, (short)0, acc2, false, false);
    acc3 = __builtin_amdgcn_wmma_f32_16x16x32_bf16(false, a.v, false, bf[3].v, (short)0, acc3, false, false);
  }
  // C/D layout: VGPR r -> row m0 + r + 8*half, col = n0 + t*16 + l16
  float* crow = C + (size_t)(m0 + half * 8) * ldc + n0 + l16;
  v8f accs[4] = {acc0, acc1, acc2, acc3};
#pragma unroll
  for (int t = 0; t < 4; ++t) {
    if (nv[t]) {
      float* cp = crow + t * 16;
#pragma unroll
      for (int rr = 0; rr < 8; ++rr) cp[(size_t)rr * ldc] = accs[t][rr];
    }
  }
}

// ----------------------------------------------- causal depthwise conv+SiLU
__global__ void conv_silu_kernel(const float* __restrict__ xz,
                                 const float* __restrict__ cw,
                                 const float* __restrict__ cb,
                                 float* __restrict__ xc,
                                 unsigned short* __restrict__ xc_bf) {
  int i = blockIdx.x * blockDim.x + threadIdx.x;
  if (i >= kTok * kDin) return;
  int d = i % kDin;
  int n = i / kDin;
  int l = n % kL;
  float w0 = cw[d * 4 + 0], w1 = cw[d * 4 + 1], w2 = cw[d * 4 + 2], w3 = cw[d * 4 + 3];
  float acc = cb[d];
  if (l >= 3) acc += xz[(size_t)(n - 3) * kD2 + d] * w0;
  if (l >= 2) acc += xz[(size_t)(n - 2) * kD2 + d] * w1;
  if (l >= 1) acc += xz[(size_t)(n - 1) * kD2 + d] * w2;
  acc += xz[(size_t)n * kD2 + d] * w3;
  float s = acc / (1.f + __expf(-acc));           // silu
  xc[i] = s;
  xc_bf[i] = f32_to_bf16(s);
}

// ------------------------------------------------ dt projection + softplus
__global__ void dtproj_kernel(const float* __restrict__ xdbl,
                              const float* __restrict__ wdt,
                              const float* __restrict__ dtb,
                              float* __restrict__ dt) {
  int i = blockIdx.x * blockDim.x + threadIdx.x;
  if (i >= kTok * kDin) return;
  int d = i % kDin;
  int n = i / kDin;
  const float* xr = xdbl + (size_t)n * kXdblPad;   // first 24 cols = dt rank part
  const float* wr = wdt + (size_t)d * kDtRank;
  float acc = dtb[d];
#pragma unroll
  for (int r = 0; r < kDtRank; ++r) acc += xr[r] * wr[r];
  dt[i] = (acc > 20.f) ? acc : log1pf(__expf(acc));
}

// --------------------------------------------------------- selective scan
// One thread per (batch, channel); h[16] kept in VGPRs, dA recomputed on the fly.
__global__ void scan_kernel(const float* __restrict__ dt, const float* __restrict__ xc,
                            const float* __restrict__ xz, const float* __restrict__ xdbl,
                            const float* __restrict__ Alog, const float* __restrict__ Dp,
                            unsigned short* __restrict__ ybf) {
  int t = blockIdx.x * blockDim.x + threadIdx.x;
  if (t >= kB * kDin) return;
  int d = t % kDin, b = t / kDin;
  float Aj[kNState];
#pragma unroll
  for (int j = 0; j < kNState; ++j) Aj[j] = -__expf(Alog[(size_t)d * kNState + j]);
  float Dv = Dp[d];
  float h[kNState];
#pragma unroll
  for (int j = 0; j < kNState; ++j) h[j] = 0.f;
  int n = b * kL;
  for (int l = 0; l < kL; ++l, ++n) {
    float dtv = dt[(size_t)n * kDin + d];
    float xcv = xc[(size_t)n * kDin + d];
    float zv  = xz[(size_t)n * kD2 + kDin + d];    // z = second half of in_proj
    float Bv[16], Cv[16];
    const float4* q = (const float4*)(xdbl + (size_t)n * kXdblPad + kDtRank);
    ((float4*)Bv)[0] = q[0]; ((float4*)Bv)[1] = q[1];
    ((float4*)Bv)[2] = q[2]; ((float4*)Bv)[3] = q[3];
    ((float4*)Cv)[0] = q[4]; ((float4*)Cv)[1] = q[5];
    ((float4*)Cv)[2] = q[6]; ((float4*)Cv)[3] = q[7];
    float dtxc = dtv * xcv;
    float y = 0.f;
#pragma unroll
    for (int j = 0; j < kNState; ++j) {
      h[j] = __expf(dtv * Aj[j]) * h[j] + dtxc * Bv[j];
      y += h[j] * Cv[j];
    }
    y += xcv * Dv;
    y *= zv / (1.f + __expf(-zv));                 // * silu(z)
    ybf[(size_t)n * kDin + d] = f32_to_bf16(y);
  }
}

inline int ceil_div(int a, int b) { return (a + b - 1) / b; }

inline void launch_gemm_full(const unsigned short* A, const unsigned short* W, float* C,
                             int M, int K, int Nout, hipStream_t s) {
  int tiles = (M >> 4) * (Nout >> 6);
  wmma_gemm_kernel<false><<<ceil_div(tiles, 8), 256, 0, s>>>(A, W, C, M, K, Nout, Nout);
}

inline void launch_gemm_edge(const unsigned short* A, const unsigned short* W, float* C,
                             int M, int K, int Nout, int ldc, hipStream_t s) {
  int tiles = (M >> 4) * (ldc >> 6);
  wmma_gemm_kernel<true><<<ceil_div(tiles, 8), 256, 0, s>>>(A, W, C, M, K, Nout, ldc);
}

}  // namespace

extern "C" void kernel_launch(void* const* d_in, const int* in_sizes, int n_in,
                              void* d_out, int out_size, void* d_ws, size_t ws_size,
                              hipStream_t stream) {
  (void)in_sizes; (void)n_in; (void)out_size; (void)ws_size;
  // setup_inputs() dict order: rgb_fea, depth_fea, then params in insertion order.
  const float* rgb      = (const float*)d_in[0];
  const float* dep      = (const float*)d_in[1];
  const float* ln_w     = (const float*)d_in[2];   // [24,384]
  const float* ln_b     = (const float*)d_in[3];   // [24,384]
  const float* W_in     = (const float*)d_in[4];   // [24,1536,384]
  const float* conv_w   = (const float*)d_in[5];   // [24,768,4]
  const float* conv_b   = (const float*)d_in[6];   // [24,768]
  const float* Wx       = (const float*)d_in[7];   // [24,56,768]
  const float* Wdt      = (const float*)d_in[8];   // [24,768,24]
  const float* dt_bias  = (const float*)d_in[9];   // [24,768]
  const float* A_log    = (const float*)d_in[10];  // [24,768,16]
  const float* Dp       = (const float*)d_in[11];  // [24,768]
  const float* Wout     = (const float*)d_in[12];  // [24,384,768]
  const float* nf_rgb_w = (const float*)d_in[13];
  const float* nf_rgb_b = (const float*)d_in[14];
  const float* nf_dep_w = (const float*)d_in[15];
  const float* nf_dep_b = (const float*)d_in[16];
  float* out = (float*)d_out;

  // ---- workspace bump allocator (256B aligned)
  char* p = (char*)d_ws;
  auto alloc = [&](size_t bytes) -> char* {
    char* q = p; p += (bytes + 255) & ~(size_t)255; return q;
  };
  float* r_rgb = (float*)alloc((size_t)kTok * kEmbed * 4);
  float* r_dep = (float*)alloc((size_t)kTok * kEmbed * 4);
  float* h_rgb = (float*)alloc((size_t)kTok * kEmbed * 4);
  float* h_dep = (float*)alloc((size_t)kTok * kEmbed * 4);
  unsigned short* xn_bf = (unsigned short*)alloc((size_t)kTok * kEmbed * 2);
  float* xz   = (float*)alloc((size_t)kTok * kD2 * 4);
  float* xc   = (float*)alloc((size_t)kTok * kDin * 4);
  unsigned short* xc_bf = (unsigned short*)alloc((size_t)kTok * kDin * 2);
  float* xdbl = (float*)alloc((size_t)kTok * kXdblPad * 4);
  float* dtb  = (float*)alloc((size_t)kTok * kDin * 4);
  unsigned short* y_bf = (unsigned short*)alloc((size_t)kTok * kDin * 2);
  unsigned short* w_in_bf  = (unsigned short*)alloc((size_t)kD2 * kEmbed * 2);
  unsigned short* w_x_bf   = (unsigned short*)alloc((size_t)kXdblReal * kDin * 2);
  unsigned short* w_out_bf = (unsigned short*)alloc((size_t)kEmbed * kDin * 2);

  // ---- init carries
  init_kernel<<<ceil_div(kTok * kEmbed, 256), 256, 0, stream>>>(
      rgb, dep, h_rgb, h_dep, r_rgb, r_dep, kTok * kEmbed);

  for (int l = 0; l < kDepth; ++l) {
    // per-layer weight conversion to bf16 (shared by both streams)
    cvt_bf16_kernel<<<ceil_div(kD2 * kEmbed, 256), 256, 0, stream>>>(
        W_in + (size_t)l * kD2 * kEmbed, w_in_bf, kD2 * kEmbed);
    cvt_bf16_kernel<<<ceil_div(kXdblReal * kDin, 256), 256, 0, stream>>>(
        Wx + (size_t)l * kXdblReal * kDin, w_x_bf, kXdblReal * kDin);
    cvt_bf16_kernel<<<ceil_div(kEmbed * kDin, 256), 256, 0, stream>>>(
        Wout + (size_t)l * kEmbed * kDin, w_out_bf, kEmbed * kDin);

    float* hs[2] = {h_rgb, h_dep};
    float* rs[2] = {r_rgb, r_dep};
    for (int s = 0; s < 2; ++s) {
      float* h = hs[s];
      float* r = rs[s];
      // r += h ; xn = LN(r)
      addnorm_kernel<<<kTok, 128, 0, stream>>>(
          h, r, ln_w + (size_t)l * kEmbed, ln_b + (size_t)l * kEmbed, xn_bf);
      // xz = xn @ W_in^T   [N,1536]
      launch_gemm_full(xn_bf, w_in_bf, xz, kTok, kEmbed, kD2, stream);
      // xc = silu(causal depthwise conv(xi) + b)
      conv_silu_kernel<<<ceil_div(kTok * kDin, 256), 256, 0, stream>>>(
          xz, conv_w + (size_t)l * kDin * 4, conv_b + (size_t)l * kDin, xc, xc_bf);
      // x_dbl = xc @ Wx^T  [N,56] (padded 64, guarded)
      launch_gemm_edge(xc_bf, w_x_bf, xdbl, kTok, kDin, kXdblReal, kXdblPad, stream);
      // dt = softplus(dt_part @ Wdt^T + dt_bias)
      dtproj_kernel<<<ceil_div(kTok * kDin, 256), 256, 0, stream>>>(
          xdbl, Wdt + (size_t)l * kDin * kDtRank, dt_bias + (size_t)l * kDin, dtb);
      // selective scan -> y_bf (includes +xc*D and *silu(z))
      scan_kernel<<<ceil_div(kB * kDin, 256), 256, 0, stream>>>(
          dtb, xc, xz, xdbl, A_log + (size_t)l * kDin * kNState,
          Dp + (size_t)l * kDin, y_bf);
      // h = y @ Wout^T     [N,384]
      launch_gemm_full(y_bf, w_out_bf, h, kTok, kDin, kEmbed, stream);
    }
  }

  // final norms -> concatenated output (rgb then depth)
  final_norm_kernel<<<kTok, 128, 0, stream>>>(h_rgb, r_rgb, nf_rgb_w, nf_rgb_b, out);
  final_norm_kernel<<<kTok, 128, 0, stream>>>(h_dep, r_dep, nf_dep_w, nf_dep_b,
                                              out + (size_t)kTok * kEmbed);
}
